// SparseEncoder_85220741087714
// MI455X (gfx1250) — matrix operand order repaired and checked by reading, not compiled, not run on hardware
//
#include <hip/hip_runtime.h>
#include <hip/hip_bf16.h>

// ---------------------------------------------------------------------------
// Sparse 3D CNN encoder (MinkowskiEngine-style) for gfx1250 / MI455X.
//
// d_in layout (setup_inputs dict order, pytrees flattened):
//   0..7   : idx_s0,msk_s0, idx_s1,msk_s1, idx_s2,msk_s2, idx_s3,msk_s3
//   8..13  : idx_d0,msk_d0, idx_d1,msk_d1, idx_d2,msk_d2
//   14..61 : 16 convs x (W f32 [27,Cin,Cout], gamma f32 [Cout], beta f32 [Cout])
//   62     : feats f32 [N0, 3]
// msk_* are numpy bool -> 1 byte each.
// ---------------------------------------------------------------------------

typedef __attribute__((ext_vector_type(16))) __bf16 v16bf;
typedef __attribute__((ext_vector_type(8)))  float  v8f;
typedef __attribute__((ext_vector_type(4)))  int    v4i;

// CDNA5 async global->LDS DMA (ASYNCcnt-tracked), with sync fallback.
// Prototype (from clang diagnostics): (v4i AS1*, v4i AS3*, imm offset, imm cpol)
#if defined(__AMDGCN__) && __has_builtin(__builtin_amdgcn_global_load_async_to_lds_b128)
#define HAVE_ASYNC 1
#else
#define HAVE_ASYNC 0
#endif

#if HAVE_ASYNC
typedef __attribute__((address_space(1))) v4i gv4i_t;
typedef __attribute__((address_space(3))) v4i lv4i_t;
#define ASYNC_CP_B128(gp, lp)                                                  \
  __builtin_amdgcn_global_load_async_to_lds_b128((gv4i_t*)(gp), (lv4i_t*)(lp), \
                                                 0, 0)
#if __has_builtin(__builtin_amdgcn_s_wait_asynccnt)
#define ASYNC_WAIT(n) __builtin_amdgcn_s_wait_asynccnt(n)
#else
#define ASYNC_WAIT(n) asm volatile("s_wait_asynccnt %0" ::"n"(n) : "memory")
#endif
#else
#define ASYNC_WAIT(n)
#endif

__device__ __forceinline__ unsigned short f32_to_bf16_rne(float f) {
  unsigned int u = __float_as_uint(f);
  unsigned int r = u + 0x7FFFu + ((u >> 16) & 1u);   // round-to-nearest-even
  return (unsigned short)(r >> 16);
}

// ---------------------------------------------------------------------------
// Weight f32 [27,Cin,Cout] -> bf16 fragment-native layout:
//   Wb[ ((k*nCC + cc)*nCG + cg)*512 + n*32 + kh*16 + j ]
//     = W[k][cc*32 + kh*16 + j][cg*16 + n]            (0 when cin >= Cin)
// so that wave lane L (n = L%16, kh = L/16) reads its 16-half B fragment as
// 32 contiguous bytes (two b128 loads) straight from global memory.
// ---------------------------------------------------------------------------
__global__ void conv_w_to_frag(const float* __restrict__ W,
                               unsigned short* __restrict__ Wb,
                               int Cin, int CinPad, int Cout, long long total) {
  long long i = (long long)blockIdx.x * blockDim.x + threadIdx.x;
  if (i >= total) return;
  int nCG = Cout >> 4;
  int j   = (int)(i & 15);
  int kh  = (int)((i >> 4) & 1);
  int n   = (int)((i >> 5) & 15);
  long long r = i >> 9;                 // ((k*nCC + cc)*nCG + cg)
  int cg  = (int)(r % nCG);
  long long r2 = r / nCG;
  int nCC = CinPad >> 5;
  int cc  = (int)(r2 % nCC);
  int k   = (int)(r2 / nCC);
  int cin  = cc * 32 + kh * 16 + j;
  int cout = cg * 16 + n;
  float v = (cin < Cin) ? W[((long long)k * Cin + cin) * Cout + cout] : 0.0f;
  Wb[i] = f32_to_bf16_rne(v);
}

// feats f32 [N,3] -> bf16 [N,32] zero-padded
__global__ void feats_to_bf16(const float* __restrict__ f,
                              unsigned short* __restrict__ xb, long long total) {
  long long i = (long long)blockIdx.x * blockDim.x + threadIdx.x;
  if (i >= total) return;
  int c = (int)(i % 32);
  long long n = i / 32;
  xb[i] = (c < 3) ? f32_to_bf16_rne(f[n * 3 + c]) : (unsigned short)0;
}

// zero the pad row (row index Nin) of a bf16 activation buffer
__global__ void zero_pad_row(unsigned short* __restrict__ xb, long long base, int C) {
  int i = blockIdx.x * blockDim.x + threadIdx.x;
  if (i < C) xb[base + i] = 0;
}

// ---------------------------------------------------------------------------
// Gather-GEMM sparse conv:  y[Nout,Cout] = sum_k gather(x, idx_k, msk_k) @ W_k
// Block: 128 threads (4 waves). Block tile: 64 sites x 64 output channels.
// Wave w owns rows [tile0+16w, +16) x all 64 cols -> 4 accumulators, so each
// A fragment feeds 4 back-to-back v_wmma_f32_16x16x32_bf16.
// A tiles: double-buffered LDS filled by GLOBAL_LOAD_ASYNC_TO_LDS_B128 with a
//          single barrier per chunk; the DMA for chunk cc+1 runs underneath
//          chunk cc's WMMAs. Masked rows gather a zeroed pad row (index Nin),
//          so the DMA is unconditional.
// B fragments load directly from global (fragment-native weight layout).
// Per-tap skip: if all 64 gathered rows are masked for offset k, skip it.
// ---------------------------------------------------------------------------
__global__ __launch_bounds__(128)
void sparse_conv_wmma(const unsigned short* __restrict__ xbf,  // [Nin+1, Cin] bf16
                      const unsigned short* __restrict__ Wbf,  // frag layout
                      const int* __restrict__ idx,             // [27, Nout]
                      const unsigned char* __restrict__ msk,   // [27, Nout] bool
                      float* __restrict__ y,                   // [Nout, Cout]
                      int Nout, int Nin, int Cin, int Cout) {
  __shared__ __align__(16) unsigned short ldsA[2][64][32];   // double-buffered
  __shared__ int srcRow[64];

  const int t     = threadIdx.x;
  const int wave  = t >> 5;                 // 0..3 -> 16-row slice
  const int lane  = t & 31;
  const int tile0 = blockIdx.x * 64;
  const int nCC   = Cin >> 5;               // K chunks (Cin multiple of 32)
  const int nCG   = Cout >> 4;              // 16-col groups in full matrix
  const int cgBase = blockIdx.y * 4;        // this block's first col group
  const int colBase = blockIdx.y * 64;

  const int m    = lane & 15;
  const int half = lane >> 4;
  const int gr   = t >> 1;                  // gather row (0..63)
  const int gseg = t & 1;                   // gather segment: halves [0,16)/[16,32)

  v8f acc0 = {}, acc1 = {}, acc2 = {}, acc3 = {};

  // issue the (async) copy of one 64x32 A chunk into LDS buffer `buf`
  auto issueA = [&](int buf, int cc) {
    int src = srcRow[gr];
    long long row = (src >= 0) ? (long long)src : (long long)Nin;  // pad row
    const unsigned short* gp = xbf + (size_t)row * Cin + (cc << 5) + (gseg << 4);
    unsigned short* lp = &ldsA[buf][gr][gseg << 4];
#if HAVE_ASYNC
    ASYNC_CP_B128(gp, lp);
    ASYNC_CP_B128(gp + 8, lp + 8);
#else
    uint4 v0 = *(const uint4*)gp;
    uint4 v1 = *(const uint4*)(gp + 8);
    *(uint4*)lp = v0;
    *(uint4*)(lp + 8) = v1;
#endif
  };

  for (int k = 0; k < 27; ++k) {
    int v = -1;
    if (t < 64) {
      int s = tile0 + t;
      if (s < Nout) {
        long long o = (long long)k * Nout + s;
        if (msk[o]) v = idx[o];
      }
      srcRow[t] = v;
    }
    // barrier + block-wide OR: publishes srcRow, fences LDS reuse, tap-skip
    if (!__syncthreads_or(v >= 0)) continue;

    // per-lane B fragment base for this tap (n*32 + kh*16 halves within group)
    const unsigned short* wkTap =
        Wbf + (((size_t)k * nCC) * nCG + cgBase) * 512 + m * 32 + half * 16;

    issueA(0, 0);                                 // prologue: chunk 0 -> buf 0

    #pragma unroll 2
    for (int cc = 0; cc < nCC; ++cc) {
      const int cur = cc & 1;
      ASYNC_WAIT(0);        // own chunk-cur DMAs complete (issued last iter)
      __syncthreads();      // join: all waves' chunk-cur DMAs landed; also
                            // proves all prior reads of buf[cur^1] consumed
      if (cc + 1 < nCC) issueA(cur ^ 1, cc + 1);  // DMA overlaps WMMAs below
      // --- A fragment (contiguous k order: k = 16*half + s, matches B) ---
      union { v16bf v; uint4 q[2]; } af;
      af.q[0] = *(const uint4*)&ldsA[cur][(wave << 4) + m][half * 16];
      af.q[1] = *(const uint4*)&ldsA[cur][(wave << 4) + m][half * 16 + 8];
      // --- 4 B fragments straight from global (32B contiguous per lane) ---
      const unsigned short* wk = wkTap + (size_t)cc * nCG * 512;
      union { v16bf v; uint4 q[2]; } b0, b1, b2, b3;
      b0.q[0] = *(const uint4*)(wk);          b0.q[1] = *(const uint4*)(wk + 8);
      b1.q[0] = *(const uint4*)(wk + 512);    b1.q[1] = *(const uint4*)(wk + 520);
      b2.q[0] = *(const uint4*)(wk + 1024);   b2.q[1] = *(const uint4*)(wk + 1032);
      b3.q[0] = *(const uint4*)(wk + 1536);   b3.q[1] = *(const uint4*)(wk + 1544);
      acc0 = __builtin_amdgcn_wmma_f32_16x16x32_bf16(false, af.v, false, b0.v,
                                                     (short)0, acc0, false, false);
      acc1 = __builtin_amdgcn_wmma_f32_16x16x32_bf16(false, af.v, false, b1.v,
                                                     (short)0, acc1, false, false);
      acc2 = __builtin_amdgcn_wmma_f32_16x16x32_bf16(false, af.v, false, b2.v,
                                                     (short)0, acc2, false, false);
      acc3 = __builtin_amdgcn_wmma_f32_16x16x32_bf16(false, af.v, false, b3.v,
                                                     (short)0, acc3, false, false);
    }
  }

  // --- store: C/D layout M = v + 8*(lane/16), N = lane%16 ---
  #pragma unroll
  for (int v = 0; v < 8; ++v) {
    int row = tile0 + (wave << 4) + v + (half << 3);
    if (row < Nout) {
      float* yr = y + (size_t)row * Cout + colBase + m;
      yr[0]  = acc0[v];
      yr[16] = acc1[v];
      yr[32] = acc2[v];
      yr[48] = acc3[v];
    }
  }
}

// ---------------------------------------------------------------------------
// BatchNorm statistics: two-stage column reduction (deterministic, no atomics)
// ---------------------------------------------------------------------------
#define NPARTS 64

__global__ __launch_bounds__(256)
void col_stats_partial(const float* __restrict__ y, float* __restrict__ part,
                       int N, int C) {
  __shared__ float sS[256], sQ[256];
  int col   = blockIdx.x * 64 + (threadIdx.x & 63);
  int rlane = threadIdx.x >> 6;                        // 0..3
  int rowsPer = (N + NPARTS - 1) / NPARTS;
  int r0 = blockIdx.y * rowsPer;
  int r1 = min(N, r0 + rowsPer);
  float s = 0.f, q = 0.f;
  for (int r = r0 + rlane; r < r1; r += 4) {
    float v = y[(size_t)r * C + col];
    s += v; q += v * v;
  }
  sS[threadIdx.x] = s; sQ[threadIdx.x] = q;
  __syncthreads();
  if (threadIdx.x < 64) {
    float ts = sS[threadIdx.x] + sS[threadIdx.x + 64] +
               sS[threadIdx.x + 128] + sS[threadIdx.x + 192];
    float tq = sQ[threadIdx.x] + sQ[threadIdx.x + 64] +
               sQ[threadIdx.x + 128] + sQ[threadIdx.x + 192];
    size_t o = ((size_t)blockIdx.y * C + col) * 2;
    part[o] = ts; part[o + 1] = tq;
  }
}

__global__ void col_stats_final(const float* __restrict__ part,
                                float* __restrict__ mu, float* __restrict__ rsig,
                                int N, int C) {
  int col = blockIdx.x * blockDim.x + threadIdx.x;
  if (col >= C) return;
  float s = 0.f, q = 0.f;
  for (int p = 0; p < NPARTS; ++p) {
    size_t o = ((size_t)p * C + col) * 2;
    s += part[o]; q += part[o + 1];
  }
  float m = s / (float)N;
  float var = q / (float)N - m * m;
  mu[col] = m;
  rsig[col] = rsqrtf(var + 1e-5f);
}

// ---------------------------------------------------------------------------
// Fused BN + exact GELU (+ optional residual); emits f32 and/or bf16 copies
// ---------------------------------------------------------------------------
__global__ void bn_gelu(const float* __restrict__ y,
                        const float* __restrict__ mu, const float* __restrict__ rsig,
                        const float* __restrict__ g, const float* __restrict__ b,
                        const float* __restrict__ res,
                        float* __restrict__ outF, unsigned short* __restrict__ outB,
                        long long total, int C) {
  long long i = (long long)blockIdx.x * blockDim.x + threadIdx.x;
  if (i >= total) return;
  int col = (int)(i % C);
  float v = (y[i] - mu[col]) * rsig[col] * g[col] + b[col];
  float t = 0.5f * v * (1.0f + erff(v * 0.70710678118654752f));  // exact GELU
  if (res)  t += res[i];
  if (outF) outF[i] = t;
  if (outB) outB[i] = f32_to_bf16_rne(t);
}

// ---------------------------------------------------------------------------
// Host orchestration
// ---------------------------------------------------------------------------
extern "C" void kernel_launch(void* const* d_in, const int* in_sizes, int n_in,
                              void* d_out, int out_size, void* d_ws, size_t ws_size,
                              hipStream_t stream) {
  const int N0 = in_sizes[0] / 27;
  const int N1 = in_sizes[2] / 27;
  const int N2 = in_sizes[4] / 27;
  const int N3 = in_sizes[6] / 27;

  // map table: 0..3 = s0..s3 ; 4..6 = d0..d2
  const int* idxArr[7]; const unsigned char* mskArr[7]; int nOutArr[7]; int nInArr[7];
  idxArr[0]=(const int*)d_in[0];  mskArr[0]=(const unsigned char*)d_in[1];  nOutArr[0]=N0; nInArr[0]=N0;
  idxArr[1]=(const int*)d_in[2];  mskArr[1]=(const unsigned char*)d_in[3];  nOutArr[1]=N1; nInArr[1]=N1;
  idxArr[2]=(const int*)d_in[4];  mskArr[2]=(const unsigned char*)d_in[5];  nOutArr[2]=N2; nInArr[2]=N2;
  idxArr[3]=(const int*)d_in[6];  mskArr[3]=(const unsigned char*)d_in[7];  nOutArr[3]=N3; nInArr[3]=N3;
  idxArr[4]=(const int*)d_in[8];  mskArr[4]=(const unsigned char*)d_in[9];  nOutArr[4]=N1; nInArr[4]=N0;
  idxArr[5]=(const int*)d_in[10]; mskArr[5]=(const unsigned char*)d_in[11]; nOutArr[5]=N2; nInArr[5]=N1;
  idxArr[6]=(const int*)d_in[12]; mskArr[6]=(const unsigned char*)d_in[13]; nOutArr[6]=N3; nInArr[6]=N2;
  const float* feats = (const float*)d_in[62];

  // 16 convs: true Cin, padded Cin, Cout, map
  static const int CIN_T[16] = {3,64,64,64,128,128,128,128,128,256,256,256,256,256,384,384};
  static const int CIN_P[16] = {32,64,64,64,128,128,128,128,128,256,256,256,256,256,384,384};
  static const int COUT [16] = {64,64,64,128,128,128,128,128,256,256,256,256,256,384,384,384};
  static const int MAP  [16] = {0,0,0,4,1,1,1,1,5,2,2,2,2,6,3,3};

  // workspace carve
  char* ws = (char*)d_ws;
  size_t off = 0;
  auto carve = [&](size_t bytes) -> char* {
    off = (off + 255) & ~(size_t)255;
    char* p = ws + off; off += bytes; return p;
  };
  unsigned short* Wb[16];
  for (int j = 0; j < 16; ++j)
    Wb[j] = (unsigned short*)carve((size_t)27 * CIN_P[j] * COUT[j] * 2);

  size_t maxNC = (size_t)N0 * 64;
  if ((size_t)N1 * 128 > maxNC) maxNC = (size_t)N1 * 128;
  if ((size_t)N2 * 256 > maxNC) maxNC = (size_t)N2 * 256;
  if ((size_t)N3 * 384 > maxNC) maxNC = (size_t)N3 * 384;
  size_t actElems = maxNC + 512;                 // + pad row (any C <= 512)

  float*          yBuf = (float*)carve(maxNC * 4);
  float*          xf   = (float*)carve(maxNC * 4);
  unsigned short* xbf  = (unsigned short*)carve(actElems * 2);
  unsigned short* hbf  = (unsigned short*)carve(actElems * 2);
  float*          part = (float*)carve((size_t)NPARTS * 384 * 2 * 4);
  float*          mu   = (float*)carve(384 * 4);
  float*          rsig = (float*)carve(384 * 4);

  // weight + feats precision/layout conversion
  for (int j = 0; j < 16; ++j) {
    long long tot = 27LL * CIN_P[j] * COUT[j];
    conv_w_to_frag<<<(unsigned)((tot + 255) / 256), 256, 0, stream>>>(
        (const float*)d_in[14 + 3 * j], Wb[j], CIN_T[j], CIN_P[j], COUT[j], tot);
  }
  {
    long long tot = (long long)N0 * 32;
    feats_to_bf16<<<(unsigned)((tot + 255) / 256), 256, 0, stream>>>(feats, xbf, tot);
  }

  // conv -> stats -> BN/GELU(+res) pipeline for one block
  auto run = [&](unsigned short* xin, int conv,
                 const float* res, float* outF, unsigned short* outB) {
    int map = MAP[conv], cin = CIN_P[conv], cout = COUT[conv];
    int Nout = nOutArr[map], Nin = nInArr[map];
    zero_pad_row<<<(unsigned)((cin + 255) / 256), 256, 0, stream>>>(
        xin, (long long)Nin * cin, cin);
    dim3 gc((unsigned)((Nout + 63) / 64), (unsigned)(cout / 64));
    sparse_conv_wmma<<<gc, 128, 0, stream>>>(
        xin, Wb[conv], idxArr[map], mskArr[map], yBuf, Nout, Nin, cin, cout);
    dim3 gs((unsigned)(cout / 64), NPARTS);
    col_stats_partial<<<gs, 256, 0, stream>>>(yBuf, part, Nout, cout);
    col_stats_final<<<(unsigned)((cout + 255) / 256), 256, 0, stream>>>(
        part, mu, rsig, Nout, cout);
    long long tot = (long long)Nout * cout;
    bn_gelu<<<(unsigned)((tot + 255) / 256), 256, 0, stream>>>(
        yBuf, mu, rsig, (const float*)d_in[15 + 3 * conv],
        (const float*)d_in[16 + 3 * conv], res, outF, outB, tot, cout);
  };

  // stem
  run(xbf, 0, nullptr, xf, xbf);
  // level0 (1 residual block)
  run(xbf, 1, nullptr, nullptr, hbf);
  run(hbf, 2, xf, xf, xbf);
  // down1
  run(xbf, 3, nullptr, xf, xbf);
  // level1 (2 residual blocks)
  run(xbf, 4, nullptr, nullptr, hbf);
  run(hbf, 5, xf, xf, xbf);
  run(xbf, 6, nullptr, nullptr, hbf);
  run(hbf, 7, xf, xf, xbf);
  // down2
  run(xbf, 8, nullptr, xf, xbf);
  // level2 (2 residual blocks)
  run(xbf, 9,  nullptr, nullptr, hbf);
  run(hbf, 10, xf, xf, xbf);
  run(xbf, 11, nullptr, nullptr, hbf);
  run(hbf, 12, xf, xf, xbf);
  // down3
  run(xbf, 13, nullptr, xf, xbf);
  // level3 (1 residual block) -> final output straight to d_out (f32)
  run(xbf, 14, nullptr, nullptr, hbf);
  run(hbf, 15, xf, (float*)d_out, nullptr);
}
